// MeshAttention_77120432767766
// MI455X (gfx1250) — compile-verified, matrix operand order checked by reading, not run on hardware
//
#include <hip/hip_runtime.h>
#include <hip/hip_bf16.h>

typedef __attribute__((ext_vector_type(16))) _Float16 v16h;
typedef __attribute__((ext_vector_type(8)))  _Float16 v8h;
typedef __attribute__((ext_vector_type(8)))  float    v8f;
typedef __attribute__((ext_vector_type(4))) unsigned int v4u;
typedef __attribute__((ext_vector_type(8)))  int v8i;
typedef __attribute__((ext_vector_type(4)))  int v4i;

#define WMMA16(a,b,c) __builtin_amdgcn_wmma_f32_16x16x32_f16(false,(a),false,(b),(short)0,(c),false,false)

// ---------------------------------------------------------------------------
// Fragment loaders following the CDNA5 16-bit WMMA VGPR layouts (ISA 7.12.2):
// A (16x32, MxK): lanes 0-15 hold M=lane, K={k0..k0+7, k0+16..k0+23};
//                 lanes 16-31 hold M=lane-16, K shifted by +8.
// B (32x16, KxN): mirrored with N=lane%16 and the same K-chunk pattern,
//                 so a [N,K] row-major (pre-transposed) operand loads as two
//                 contiguous 16-byte chunks per lane.
// C/D (16x16 f32): lane n=lane%16 holds rows m = i + 8*(lane/16), i=0..7.
// ---------------------------------------------------------------------------
__device__ __forceinline__ v16h load_frag16(const _Float16* __restrict__ base,
                                            int ld, int row0, int k0) {
  const int lane = threadIdx.x & 31;
  const int r = row0 + (lane & 15);
  const int hh = lane >> 4;
  const _Float16* p0 = base + (size_t)r * ld + k0 + 8 * hh;
  v8h a = *(const v8h*)p0;
  v8h b = *(const v8h*)(p0 + 16);
  v16h out;
#pragma unroll
  for (int i = 0; i < 8; i++) { out[i] = a[i]; out[i + 8] = b[i]; }
  return out;
}

// Same mapping but source is f32 (used to feed attn probabilities from LDS).
__device__ __forceinline__ v16h load_frag16_f32(const float* __restrict__ base,
                                                int ld, int row0, int k0) {
  const int lane = threadIdx.x & 31;
  const int r = row0 + (lane & 15);
  const int hh = lane >> 4;
  const float* p0 = base + (size_t)r * ld + k0 + 8 * hh;
  v16h out;
#pragma unroll
  for (int i = 0; i < 8; i++) {
    out[i]     = (_Float16)p0[i];
    out[i + 8] = (_Float16)p0[i + 16];
  }
  return out;
}

// ---------------------------------------------------------------------------
// TDM: async 2-D tile load (rows x 32 f16, row stride 512 elems) into LDS.
// Builds D# group0/group1 per ISA 08_async_tensor 8.3/8.4:
//   g0: count=1 | lds_addr | global_addr | type=2
//   g1: data_size=1(2B), tensor_dim0=32, tensor_dim1=rows, tile_dim0=32,
//       tile_dim1=rows, tensor_dim0_stride=512
// Issued once per wave (EXEC ignored); tracked by TENSORcnt.
// This toolchain's builtin takes 6 args (extra int32x8 group before cpol).
// ---------------------------------------------------------------------------
__device__ __forceinline__ void tdm_load_tile(unsigned lds_off,
                                              const _Float16* gptr,
                                              unsigned rows) {
  unsigned long long ga = (unsigned long long)(uintptr_t)gptr;
  v4u g0;
  g0[0] = 1u;                                        // count=1, user mode
  g0[1] = lds_off;                                   // lds_addr (bytes)
  g0[2] = (unsigned)(ga & 0xFFFFFFFFu);              // global_addr[31:0]
  g0[3] = (unsigned)((ga >> 32) & 0x1FFFFFFu) | (2u << 30);  // addr[56:32]|type=2
  v8i g1;
  g1[0] = (1 << 16);                                 // data_size=1 -> 2 bytes
  g1[1] = (32 << 16);                                // tensor_dim0[15:0]=32
  g1[2] = (int)((rows & 0xFFFFu) << 16);             // dim0 hi=0 | tensor_dim1 lo=rows
  g1[3] = (32 << 16);                                // dim1 hi=0 | tile_dim0=32
  g1[4] = (int)rows;                                 // tile_dim1=rows | tile_dim2=0
  g1[5] = 512;                                       // tensor_dim0_stride lo32
  g1[6] = 0;                                         // stride hi | dim1_stride lo
  g1[7] = 0;
  v4i z4 = {0, 0, 0, 0};
  v8i z8 = {0, 0, 0, 0, 0, 0, 0, 0};
  __builtin_amdgcn_tensor_load_to_lds(g0, g1, z4, z4, z8, 0);
}

// ---------------------------------------------------------------------------
// Kernel 1: s = transpose(x), LayerNorm(s) -> qn ; both stored f16 [B*E, DM]
// ---------------------------------------------------------------------------
__global__ void prep_kernel(const float* __restrict__ x,
                            const float* __restrict__ g,
                            const float* __restrict__ bb,
                            _Float16* __restrict__ sT,
                            _Float16* __restrict__ qn) {
  const int be = blockIdx.x;                 // b*1024 + e
  const int b = be >> 10, e = be & 1023;
  __shared__ float r1[128], r2[128];
  float vals[4];
  float s = 0.f, ss = 0.f;
#pragma unroll
  for (int i = 0; i < 4; i++) {
    int d = threadIdx.x + i * 128;
    float v = x[((size_t)b * 512 + d) * 1024 + e];
    vals[i] = v; s += v; ss += v * v;
  }
  r1[threadIdx.x] = s; r2[threadIdx.x] = ss;
  __syncthreads();
  for (int st = 64; st > 0; st >>= 1) {
    if (threadIdx.x < st) {
      r1[threadIdx.x] += r1[threadIdx.x + st];
      r2[threadIdx.x] += r2[threadIdx.x + st];
    }
    __syncthreads();
  }
  float mu   = r1[0] * (1.f / 512.f);
  float var  = r2[0] * (1.f / 512.f) - mu * mu;
  float rstd = rsqrtf(var + 1e-6f);
#pragma unroll
  for (int i = 0; i < 4; i++) {
    int d = threadIdx.x + i * 128;
    size_t o = (size_t)be * 512 + d;
    sT[o] = (_Float16)vals[i];
    qn[o] = (_Float16)((vals[i] - mu) * rstd * g[d] + bb[d]);
  }
}

// ---------------------------------------------------------------------------
// Kernel 2: convert W [K=512,N=512] f32 -> Wt [N,K] f16 (transposed)
// ---------------------------------------------------------------------------
__global__ void wcvt_kernel(const float* __restrict__ W, _Float16* __restrict__ Wt) {
  int idx = blockIdx.x * 256 + threadIdx.x;    // < 262144
  int n = idx >> 9, k = idx & 511;
  Wt[idx] = (_Float16)W[(size_t)k * 512 + n];
}

// ---------------------------------------------------------------------------
// Kernel 3/6: WMMA GEMM  C[M=2048,N=512] = A[M,512] x Wt^T, scatter-store.
// TDM double-buffered LDS staging: As[2][64][32] + Bs[2][128][32] (24 KB).
// Wave 0 issues two async tile DMAs per K-step, pipelined one step ahead;
// TENSORcnt is in-order, so s_wait_tensorcnt<=2 after issuing step i+1
// guarantees step i is resident. All frag reads are ds_load_b128.
// mode 0: qs = q/8 -> [B,H,E,DK] f16     mode 1: k -> [B,H,E,DK] f16
// mode 2: v  -> vT [B,H,DV,E] f16        mode 3: fc + residual -> x_out f32
// Block = 8 waves: 4 along M x 2 along N -> 64x128 tile. Grid (32,4).
// ---------------------------------------------------------------------------
__global__ void gemm_kernel(const _Float16* __restrict__ A,
                            const _Float16* __restrict__ Wt,
                            void* __restrict__ outp,
                            const float* __restrict__ resid,
                            int mode) {
  __shared__ _Float16 As[2][64][32];    // 8 KB
  __shared__ _Float16 Bs[2][128][32];   // 16 KB
  const int w = threadIdx.x >> 5;
  const int lane = threadIdx.x & 31;
  const int wm = w & 3, wn = w >> 2;
  const int m0b = blockIdx.x * 64;
  const int n0b = blockIdx.y * 128;
  const int m0 = m0b + wm * 16;
  const int n0 = n0b + wn * 64;

  if (w == 0) {
    tdm_load_tile((unsigned)(uintptr_t)&As[0][0][0], A  + (size_t)m0b * 512, 64);
    tdm_load_tile((unsigned)(uintptr_t)&Bs[0][0][0], Wt + (size_t)n0b * 512, 128);
  }

  v8f acc[4] = {v8f{}, v8f{}, v8f{}, v8f{}};
  for (int it = 0; it < 16; ++it) {
    const int buf = it & 1;
    if (w == 0) {
      if (it + 1 < 16) {
        tdm_load_tile((unsigned)(uintptr_t)&As[buf ^ 1][0][0],
                      A  + (size_t)m0b * 512 + (it + 1) * 32, 64);
        tdm_load_tile((unsigned)(uintptr_t)&Bs[buf ^ 1][0][0],
                      Wt + (size_t)n0b * 512 + (it + 1) * 32, 128);
        __builtin_amdgcn_s_wait_tensorcnt(2);
      } else {
        __builtin_amdgcn_s_wait_tensorcnt(0);
      }
    }
    __syncthreads();
    v16h a = load_frag16(&As[buf][0][0], 32, wm * 16, 0);
#pragma unroll
    for (int f = 0; f < 4; f++) {
      v16h bf = load_frag16(&Bs[buf][0][0], 32, wn * 64 + f * 16, 0);
      acc[f] = WMMA16(a, bf, acc[f]);
    }
    __syncthreads();
  }

  const int nl = lane & 15, hh = lane >> 4;
#pragma unroll
  for (int f = 0; f < 4; f++) {
    int n = n0 + f * 16 + nl;
#pragma unroll
    for (int ii = 0; ii < 8; ii++) {
      int m = m0 + ii + 8 * hh;
      int b = m >> 10, e = m & 1023;
      float vv = acc[f][ii];
      if (mode == 0) {
        int hd = n >> 6, dk = n & 63;
        ((_Float16*)outp)[(((size_t)(b * 8 + hd)) * 1024 + e) * 64 + dk] =
            (_Float16)(vv * 0.125f);                       // / sqrt(64)
      } else if (mode == 1) {
        int hd = n >> 6, dk = n & 63;
        ((_Float16*)outp)[(((size_t)(b * 8 + hd)) * 1024 + e) * 64 + dk] = (_Float16)vv;
      } else if (mode == 2) {
        int hd = n >> 6, dv = n & 63;
        ((_Float16*)outp)[(((size_t)(b * 8 + hd)) * 64 + dv) * 1024 + e] = (_Float16)vv;
      } else {
        size_t idx = ((size_t)b * 512 + n) * 1024 + e;     // x_out[b, d=n, e]
        ((float*)outp)[idx] = vv + resid[idx];
      }
    }
  }
}

// ---------------------------------------------------------------------------
// Kernel 4: qdot[b,h,e,c] = qs[b,h,e,:] . base_rpr[c,:]   (c = 0..5)
// ---------------------------------------------------------------------------
__global__ void qdot_kernel(const _Float16* __restrict__ qs,
                            const float* __restrict__ rpr,
                            float* __restrict__ qdot) {
  int idx = blockIdx.x * 256 + threadIdx.x;   // < 98304
  int r = idx / 6, c = idx - r * 6;
  const _Float16* q = qs + (size_t)r * 64;
  const float* p = rpr + c * 64;
  float s = 0.f;
#pragma unroll 8
  for (int d = 0; d < 64; d++) s += (float)q[d] * p[d];
  qdot[(size_t)r * 8 + c] = s;
}

__global__ void zero_kernel(float* __restrict__ p, int n) {
  int i = blockIdx.x * 256 + threadIdx.x;
  if (i < n) p[i] = 0.f;
}

// ---------------------------------------------------------------------------
// Kernel 5: attention. One block per (b, h, 16-row tile). 8 waves.
// logits via WMMA + qdot gather + mask -> LDS; block softmax; write attn f32;
// column sums for attn_per_edge; attn @ V via WMMA (P read back from LDS).
// ---------------------------------------------------------------------------
__global__ void attn_kernel(const _Float16* __restrict__ qs,
                            const _Float16* __restrict__ k16,
                            const _Float16* __restrict__ vT,
                            const float* __restrict__ qdot,
                            const int* __restrict__ dist,
                            float* __restrict__ attn_out,
                            _Float16* __restrict__ out_heads,
                            float* __restrict__ acc) {
  const int bt = blockIdx.x;
  const int it = bt & 63, h = (bt >> 6) & 7, b = bt >> 9;
  const int i0 = it << 4;
  const int w = threadIdx.x >> 5;
  const int lane = threadIdx.x & 31;
  const int nl = lane & 15, hh = lane >> 4;

  __shared__ float lg[16][1024];     // logits -> probabilities (64 KB)
  __shared__ float red[16][17];
  __shared__ float rowmax[16], rowinv[16];
  __shared__ float part[16][64];

  const _Float16* qbase = qs  + ((size_t)(b * 8 + h)) * 1024 * 64;
  const _Float16* kbase = k16 + ((size_t)(b * 8 + h)) * 1024 * 64;
  const _Float16* vbase = vT  + ((size_t)(b * 8 + h)) * 64 * 1024;
  const float*    qd    = qdot + ((size_t)(b * 8 + h)) * 1024 * 8;
  const int*      drow  = dist + (size_t)b * 1024 * 1024;

  v16h a0 = load_frag16(qbase, 64, i0, 0);
  v16h a1 = load_frag16(qbase, 64, i0, 32);

  // ---- logits: each wave owns j-tiles w, w+8, ... (64 tiles total) ----
  for (int jt = w; jt < 64; jt += 8) {
    int j0 = jt << 4;
    v8f c = {};
    c = WMMA16(a0, load_frag16(kbase, 64, j0, 0),  c);
    c = WMMA16(a1, load_frag16(kbase, 64, j0, 32), c);
    int j = j0 + nl;
#pragma unroll
    for (int ii = 0; ii < 8; ii++) {
      int m = ii + 8 * hh;
      int i = i0 + m;
      int dv = drow[(size_t)i * 1024 + j];
      float val = (dv <= 5) ? (c[ii] + qd[(size_t)i * 8 + dv]) : -1e9f;
      lg[m][j] = val;
    }
  }
  __syncthreads();

  // ---- row max ----
  {
    int r = threadIdx.x & 15, ch = threadIdx.x >> 4;
    float mx = -3.4e38f;
    for (int j = ch * 64; j < ch * 64 + 64; j++) mx = fmaxf(mx, lg[r][j]);
    red[r][ch] = mx;
  }
  __syncthreads();
  if (threadIdx.x < 16) {
    float mx = -3.4e38f;
    for (int c2 = 0; c2 < 16; c2++) mx = fmaxf(mx, red[threadIdx.x][c2]);
    rowmax[threadIdx.x] = mx;
  }
  __syncthreads();

  // ---- exp + row sum ----
  {
    int r = threadIdx.x & 15, ch = threadIdx.x >> 4;
    float mx = rowmax[r], s = 0.f;
    for (int j = ch * 64; j < ch * 64 + 64; j++) {
      float p = __expf(lg[r][j] - mx);
      lg[r][j] = p; s += p;
    }
    red[r][ch] = s;
  }
  __syncthreads();
  if (threadIdx.x < 16) {
    float s = 0.f;
    for (int c2 = 0; c2 < 16; c2++) s += red[threadIdx.x][c2];
    rowinv[threadIdx.x] = 1.f / s;
  }
  __syncthreads();

  // ---- normalize + write attn (f32) ----
  float* arow = attn_out + (((size_t)(b * 8 + h)) * 1024 + i0) * 1024;
  {
    int r = threadIdx.x & 15, ch = threadIdx.x >> 4;
    float inv = rowinv[r];
    for (int j = ch * 64; j < ch * 64 + 64; j++) {
      float p = lg[r][j] * inv;
      lg[r][j] = p;
      arow[(size_t)r * 1024 + j] = p;
    }
  }
  __syncthreads();

  // ---- attn_per_edge partial: sum over this block's (h, i-tile) rows.
  // masked entries underflow to exactly 0 in exp(), so p already == p*mask.
  for (int j = threadIdx.x; j < 1024; j += 256) {
    float s = 0.f;
#pragma unroll
    for (int r = 0; r < 16; r++) s += lg[r][j];
    atomicAdd(&acc[b * 1024 + j], s);
  }

  // ---- out = P @ V : 4 dv-tiles x 2 K-halves across the 8 waves ----
  const int ntile = w & 3, khalf = w >> 2;
  v8f oc = {};
  for (int kt = khalf * 512; kt < khalf * 512 + 512; kt += 32) {
    v16h pa = load_frag16_f32(&lg[0][0], 1024, 0, kt);
    v16h vb = load_frag16(vbase, 1024, ntile * 16, kt);
    oc = WMMA16(pa, vb, oc);
  }
  if (khalf == 0) {
#pragma unroll
    for (int ii = 0; ii < 8; ii++) part[ii + 8 * hh][ntile * 16 + nl] = oc[ii];
  }
  __syncthreads();
  if (khalf == 1) {
#pragma unroll
    for (int ii = 0; ii < 8; ii++) {
      int m = ii + 8 * hh;
      float vfull = oc[ii] + part[m][ntile * 16 + nl];
      int i = i0 + m;
      out_heads[((size_t)b * 1024 + i) * 512 + h * 64 + ntile * 16 + nl] = (_Float16)vfull;
    }
  }
}

// ---------------------------------------------------------------------------
// Kernel 7: attn_per_edge[b,j] = acc[b,j] / sum_i mask[b,i,j]
// ---------------------------------------------------------------------------
__global__ void edge_kernel(const int* __restrict__ dist,
                            const float* __restrict__ acc,
                            float* __restrict__ out) {
  const int bj = blockIdx.x;
  const int b = bj >> 10, j = bj & 1023;
  __shared__ int cred[128];
  int cnt = 0;
  for (int i = threadIdx.x; i < 1024; i += 128)
    cnt += (dist[((size_t)b * 1024 + i) * 1024 + j] <= 5) ? 1 : 0;
  cred[threadIdx.x] = cnt;
  __syncthreads();
  for (int st = 64; st > 0; st >>= 1) {
    if (threadIdx.x < st) cred[threadIdx.x] += cred[threadIdx.x + st];
    __syncthreads();
  }
  if (threadIdx.x == 0) out[bj] = acc[bj] / (float)cred[0];
}

// ---------------------------------------------------------------------------
extern "C" void kernel_launch(void* const* d_in, const int* in_sizes, int n_in,
                              void* d_out, int out_size, void* d_ws, size_t ws_size,
                              hipStream_t stream) {
  const float* x        = (const float*)d_in[0];   // [B,DM,E,1]
  const int*   dist     = (const int*)  d_in[1];   // [B,E,E]
  const float* base_rpr = (const float*)d_in[2];   // [6,DK]
  const float* w_q      = (const float*)d_in[3];
  const float* w_k      = (const float*)d_in[4];
  const float* w_v      = (const float*)d_in[5];
  const float* w_fc     = (const float*)d_in[6];
  const float* ln_g     = (const float*)d_in[7];
  const float* ln_b     = (const float*)d_in[8];

  char* ws = (char*)d_ws;
  const size_t MB = 1u << 20;
  _Float16* sT   = (_Float16*)(ws + 0 * MB);          // [B*E,512] f16 (2 MB)
  _Float16* qn   = (_Float16*)(ws + 2 * MB);          // 2 MB
  _Float16* wqT  = (_Float16*)(ws + 4 * MB);          // [512,512] f16 each
  _Float16* wkT  = (_Float16*)(ws + 4 * MB + 512 * 1024);
  _Float16* wvT  = (_Float16*)(ws + 5 * MB);
  _Float16* wfcT = (_Float16*)(ws + 5 * MB + 512 * 1024);
  _Float16* qsx  = (_Float16*)(ws + 6 * MB);          // [B,H,E,64]
  _Float16* k16  = (_Float16*)(ws + 8 * MB);          // [B,H,E,64]
  _Float16* vT   = (_Float16*)(ws + 10 * MB);         // [B,H,64,E]
  _Float16* outh = (_Float16*)(ws + 12 * MB);         // [B,E,512]
  float*    qdot = (float*)   (ws + 14 * MB);         // [B*H*E, 8]
  float*    acc  = (float*)   (ws + 14 * MB + 512 * 1024);  // [B,E]

  float* xout = (float*)d_out;                         // [B,DM,E]  1,048,576
  float* attn = xout + (size_t)2 * 512 * 1024;         // [B,H,E,E] 16,777,216
  float* ape  = attn + (size_t)2 * 8 * 1024 * 1024;    // [B,E]     2,048

  prep_kernel<<<2048, 128, 0, stream>>>(x, ln_g, ln_b, sT, qn);

  wcvt_kernel<<<1024, 256, 0, stream>>>(w_q,  wqT);
  wcvt_kernel<<<1024, 256, 0, stream>>>(w_k,  wkT);
  wcvt_kernel<<<1024, 256, 0, stream>>>(w_v,  wvT);
  wcvt_kernel<<<1024, 256, 0, stream>>>(w_fc, wfcT);

  gemm_kernel<<<dim3(32, 4), 256, 0, stream>>>(qn, wqT, (void*)qsx, nullptr, 0);
  gemm_kernel<<<dim3(32, 4), 256, 0, stream>>>(sT, wkT, (void*)k16, nullptr, 1);
  gemm_kernel<<<dim3(32, 4), 256, 0, stream>>>(sT, wvT, (void*)vT,  nullptr, 2);

  qdot_kernel<<<384, 256, 0, stream>>>(qsx, base_rpr, qdot);
  zero_kernel<<<8, 256, 0, stream>>>(acc, 2048);

  attn_kernel<<<1024, 256, 0, stream>>>(qsx, k16, vT, qdot, dist, attn, outh, acc);

  gemm_kernel<<<dim3(32, 4), 256, 0, stream>>>(outh, wfcT, (void*)xout, x, 3);

  edge_kernel<<<2048, 128, 0, stream>>>(dist, acc, ape);
}
